// ESNLanguageModel_25443386261521
// MI455X (gfx1250) — matrix operand order repaired and checked by reading, not compiled, not run on hardware
//
#include <hip/hip_runtime.h>
#include <cmath>

// ---------------------------------------------------------------------------
// Types
// ---------------------------------------------------------------------------
typedef __attribute__((ext_vector_type(16))) __bf16 v16bf;
typedef __attribute__((ext_vector_type(8)))  float  v8f;

struct alignas(16) U4 { unsigned x, y, z, w; };

union Frag {
    v16bf v;
    U4    u[2];
};

// fp32 -> bf16 round-to-nearest-even
__device__ __forceinline__ unsigned short f2bf(float f) {
    unsigned u = __builtin_bit_cast(unsigned, f);
    u += 0x7FFFu + ((u >> 16) & 1u);
    return (unsigned short)(u >> 16);
}
__device__ __forceinline__ unsigned pk2(float a, float b) {
    return (unsigned)f2bf(a) | ((unsigned)f2bf(b) << 16);
}

// ---------------------------------------------------------------------------
// Prep kernels (one-time fp32 -> bf16 conversions)
// ---------------------------------------------------------------------------

// emb_bf16[m][d] = bf16(tok_emb[x[m]][d]),  m in [0,2048), d in [0,512)
__global__ __launch_bounds__(128)
void gather_convert_emb(const int* __restrict__ x, const float* __restrict__ tok_emb,
                        unsigned short* __restrict__ out)
{
    const int m = blockIdx.x;
    const int d = threadIdx.x * 4;
    const float4 f = *(const float4*)(tok_emb + (long)x[m] * 512 + d);
    uint2 o; o.x = pk2(f.x, f.y); o.y = pk2(f.z, f.w);
    *(uint2*)(out + (long)m * 512 + d) = o;
}

// flat fp32 -> bf16 (same layout)
__global__ __launch_bounds__(256)
void convert_f32_bf16(const float* __restrict__ in, unsigned short* __restrict__ out, long n)
{
    const long i = ((long)blockIdx.x * blockDim.x + threadIdx.x) * 4;
    if (i + 3 < n) {
        const float4 f = *(const float4*)(in + i);
        uint2 o; o.x = pk2(f.x, f.y); o.y = pk2(f.z, f.w);
        *(uint2*)(out + i) = o;
    } else {
        for (long j = i; j < n; ++j) out[j] = f2bf(in[j]);
    }
}

// W_in [512][2048] fp32 -> W_inT [2048][512] bf16 (so the ep GEMM is NT)
__global__ __launch_bounds__(256)
void transpose_convert_512x2048(const float* __restrict__ in, unsigned short* __restrict__ out)
{
    __shared__ float tile[32][33];
    const int tx = threadIdx.x, ty = threadIdx.y;
    const int bx = blockIdx.x * 32;   // col of in (0..2047)
    const int by = blockIdx.y * 32;   // row of in (0..511)
#pragma unroll
    for (int r = 0; r < 32; r += 8)
        tile[ty + r][tx] = in[(long)(by + ty + r) * 2048 + bx + tx];
    __syncthreads();
#pragma unroll
    for (int r = 0; r < 32; r += 8)
        out[(long)(bx + ty + r) * 512 + by + tx] = f2bf(tile[tx][ty + r]);
}

// Init the double-buffered h (16 rows x 2048, rows 8..15 are zero padding)
// and the grid-barrier counter. h_init[b][j] = h0[j] for all batches b.
__global__ __launch_bounds__(256)
void scan_init(const float* __restrict__ h0, unsigned short* __restrict__ hb0,
               unsigned short* __restrict__ hb1, unsigned* __restrict__ cnt)
{
    const int i    = blockIdx.x * 256 + threadIdx.x;   // 0..32767
    const int row  = i >> 11;
    const int colj = i & 2047;
    const unsigned short v = (row < 8) ? f2bf(h0[colj]) : (unsigned short)0;
    hb0[i] = v;
    hb1[i] = v;
    if (i == 0) *cnt = 0u;
}

// ---------------------------------------------------------------------------
// WMMA GEMM (all-NT, bf16 operands):  C[M,N] = A[M,K] * B[N,K]^T (+bias)
// Double-buffered LDS staging via GLOBAL_LOAD_ASYNC_TO_LDS_B128.
// ---------------------------------------------------------------------------
#define BM 128
#define BN 128
#define BK 32
#define LDSS 48   // LDS row stride in bf16 units (96B: 16B aligned, bank-staggered)

template<bool HAS_BIAS, bool OUT_BF16>
__global__ __launch_bounds__(256)
void gemm_bf16_nt_async(const unsigned short* __restrict__ A, int lda,
                        const unsigned short* __restrict__ B, int ldb,
                        const float* __restrict__ bias,
                        void* __restrict__ Cv, int ldc,
                        int M, int N, int K)
{
    __shared__ unsigned short As[2][BM * LDSS];
    __shared__ unsigned short Bs[2][BN * LDSS];

    const int tid  = threadIdx.x;
    const int lane = tid & 31;
    const int wave = tid >> 5;
    const int half = lane >> 4;
    const int l16  = lane & 15;
    const int wm   = wave >> 2;   // 0..1  -> 64 rows
    const int wn   = wave & 3;    // 0..3  -> 32 cols
    const int m0   = blockIdx.y * BM;
    const int n0   = blockIdx.x * BN;

    const unsigned asB[2] = { (unsigned)(uintptr_t)(&As[0][0]), (unsigned)(uintptr_t)(&As[1][0]) };
    const unsigned bsB[2] = { (unsigned)(uintptr_t)(&Bs[0][0]), (unsigned)(uintptr_t)(&Bs[1][0]) };

    // 128x32 bf16 tile = 512 x 16B chunks; 2 chunks per thread
    const int c0   = tid * 2;
    const int row0 = c0 >> 2;
    const int sub0 = (c0 & 3) * 16;
    const int row1 = (c0 + 1) >> 2;
    const int sub1 = ((c0 + 1) & 3) * 16;

    auto issue = [&](int buf, int k0) {
        {
            const unsigned long long ga0 =
                (unsigned long long)(uintptr_t)((const char*)(A + (long)(m0 + row0) * lda + k0) + sub0);
            const unsigned la0 = asB[buf] + row0 * (LDSS * 2) + sub0;
            asm volatile("global_load_async_to_lds_b128 %0, %1, off" :: "v"(la0), "v"(ga0) : "memory");
            const unsigned long long ga1 =
                (unsigned long long)(uintptr_t)((const char*)(A + (long)(m0 + row1) * lda + k0) + sub1);
            const unsigned la1 = asB[buf] + row1 * (LDSS * 2) + sub1;
            asm volatile("global_load_async_to_lds_b128 %0, %1, off" :: "v"(la1), "v"(ga1) : "memory");
        }
        if (n0 + row0 < N) {
            const unsigned long long gb0 =
                (unsigned long long)(uintptr_t)((const char*)(B + (long)(n0 + row0) * ldb + k0) + sub0);
            const unsigned lb0 = bsB[buf] + row0 * (LDSS * 2) + sub0;
            asm volatile("global_load_async_to_lds_b128 %0, %1, off" :: "v"(lb0), "v"(gb0) : "memory");
        }
        if (n0 + row1 < N) {
            const unsigned long long gb1 =
                (unsigned long long)(uintptr_t)((const char*)(B + (long)(n0 + row1) * ldb + k0) + sub1);
            const unsigned lb1 = bsB[buf] + row1 * (LDSS * 2) + sub1;
            asm volatile("global_load_async_to_lds_b128 %0, %1, off" :: "v"(lb1), "v"(gb1) : "memory");
        }
    };

    v8f acc[4][2];
#pragma unroll
    for (int i = 0; i < 4; ++i)
#pragma unroll
        for (int j = 0; j < 2; ++j)
#pragma unroll
            for (int e = 0; e < 8; ++e) acc[i][j][e] = 0.0f;

    issue(0, 0);
    int cur = 0;

    for (int k0 = 0; k0 < K; k0 += BK) {
        asm volatile("s_wait_asynccnt 0x0" ::: "memory");
        __syncthreads();
        if (k0 + BK < K) issue(cur ^ 1, k0 + BK);

        Frag af[4];
#pragma unroll
        for (int mt = 0; mt < 4; ++mt) {
            const int row = wm * 64 + mt * 16 + l16;
            const unsigned short* p = &As[cur][row * LDSS + half * 8];
            af[mt].u[0] = *(const U4*)(p);
            af[mt].u[1] = *(const U4*)(p + 16);
        }
        Frag bf[2];
#pragma unroll
        for (int nt = 0; nt < 2; ++nt) {
            const int col = wn * 32 + nt * 16 + l16;
            const unsigned short* p = &Bs[cur][col * LDSS + half * 8];
            bf[nt].u[0] = *(const U4*)(p);
            bf[nt].u[1] = *(const U4*)(p + 16);
        }

#pragma unroll
        for (int mt = 0; mt < 4; ++mt)
#pragma unroll
            for (int nt = 0; nt < 2; ++nt)
                acc[mt][nt] = __builtin_amdgcn_wmma_f32_16x16x32_bf16(
                    false, af[mt].v, false, bf[nt].v,
                    (short)0, acc[mt][nt], false, false);

        __syncthreads();
        cur ^= 1;
    }

#pragma unroll
    for (int mt = 0; mt < 4; ++mt) {
        const int rbase = m0 + wm * 64 + mt * 16 + half * 8;
#pragma unroll
        for (int nt = 0; nt < 2; ++nt) {
            const int col = n0 + wn * 32 + nt * 16 + l16;
            if (col < N) {
                const float bv = HAS_BIAS ? bias[col] : 0.0f;
#pragma unroll
                for (int r = 0; r < 8; ++r) {
                    const float v = acc[mt][nt][r] + bv;
                    if (OUT_BF16)
                        ((unsigned short*)Cv)[(long)(rbase + r) * ldc + col] = f2bf(v);
                    else
                        ((float*)Cv)[(long)(rbase + r) * ldc + col] = v;
                }
            }
        }
    }
}

// ---------------------------------------------------------------------------
// Grid-wide software barrier (monotonic counter, 64 arrivals per phase).
// ---------------------------------------------------------------------------
__device__ __forceinline__ void grid_barrier(unsigned* cnt, unsigned target) {
    __threadfence();                 // release this WG's global writes
    __syncthreads();
    if (threadIdx.x == 0) {
        __hip_atomic_fetch_add(cnt, 1u, __ATOMIC_ACQ_REL, __HIP_MEMORY_SCOPE_AGENT);
        while (__hip_atomic_load(cnt, __ATOMIC_ACQUIRE, __HIP_MEMORY_SCOPE_AGENT) < target)
            __builtin_amdgcn_s_sleep(2);
    }
    __syncthreads();
    __threadfence();                 // acquire: drop stale cached h lines
}

// ---------------------------------------------------------------------------
// WMMA reservoir scan, 64 persistent workgroups.
//   - WG g owns output columns [g*32, g*32+32).
//   - Its 32 rows of W_rec (bf16, NT layout) are staged ONCE into LDS via
//     async-to-LDS (131.6 KB) and reused for all 256 steps.
//   - Per step: rec[16x32] = h[16x2048] @ Wslab^T via v_wmma (8 waves split K,
//     2 N-tiles each), partials reduced in LDS, fp32 tanh/leak update,
//     new h broadcast through a double-buffered 16x2048 bf16 global buffer,
//     one grid barrier per step.
// Rows 8..15 of h are zero padding (M=8 batches padded to WMMA M=16).
// ---------------------------------------------------------------------------
#define SCAN_WGS 64
#define KP 2056   // LDS row stride (bf16) for the W slab: 4112B rows, bank-staggered

__global__ __launch_bounds__(256)
void esn_scan_wmma(const float* __restrict__ ep, const unsigned short* __restrict__ Wrec16,
                   const float* __restrict__ a, const float* __restrict__ h0,
                   unsigned short* __restrict__ hb0, unsigned short* __restrict__ hb1,
                   unsigned short* __restrict__ hs16, unsigned* __restrict__ cnt)
{
    extern __shared__ char smem[];
    unsigned short* Bsl  = (unsigned short*)smem;                    // 32 x KP bf16
    float*          part = (float*)(smem + (long)32 * KP * 2);       // 8 waves x 512 f32

    const int g    = blockIdx.x;          // 0..63
    const int tid  = threadIdx.x;
    const int lane = tid & 31;
    const int wave = tid >> 5;
    const int half = lane >> 4;
    const int l16  = lane & 15;

    // ---- stage my 32 W_rec rows into LDS once (async-to-LDS, 8192 chunks) ----
    const unsigned bslBase = (unsigned)(uintptr_t)(&Bsl[0]);
#pragma unroll 4
    for (int q = 0; q < 32; ++q) {
        const unsigned long long ga = (unsigned long long)(uintptr_t)
            ((const char*)(Wrec16 + (long)(g * 32 + q) * 2048) + tid * 16);
        const unsigned la = bslBase + q * (KP * 2) + tid * 16;
        asm volatile("global_load_async_to_lds_b128 %0, %1, off" :: "v"(la), "v"(ga) : "memory");
    }
    asm volatile("s_wait_asynccnt 0x0" ::: "memory");
    __syncthreads();

    // update-phase ownership: thread -> (batch ub, local col ui)
    const int ub  = tid >> 5;             // 0..7
    const int ui  = tid & 31;             // 0..31
    const int col = g * 32 + ui;
    const float aa   = a[col];
    float       hcur = h0[col];           // h_init is h0 broadcast over batches

    for (int t = 0; t < 256; ++t) {
        const unsigned short* hc = (t & 1) ? hb1 : hb0;
        unsigned short*       hn = (t & 1) ? hb0 : hb1;

        // ---- wave-level WMMA over K slice [wave*256, wave*256+256) ----
        v8f acc0, acc1;
#pragma unroll
        for (int e = 0; e < 8; ++e) { acc0[e] = 0.0f; acc1[e] = 0.0f; }
        const int ks0 = wave * 256;
#pragma unroll
        for (int kk = 0; kk < 8; ++kk) {
            const int k0 = ks0 + kk * 32;
            Frag af;
            const U4* pa = (const U4*)(hc + (long)l16 * 2048 + k0 + half * 8);
            af.u[0] = pa[0];
            af.u[1] = pa[2];          // K offset +16 elements = +32B
            Frag b0, b1;
            const unsigned short* pb0 = &Bsl[l16 * KP + k0 + half * 8];
            b0.u[0] = *(const U4*)(pb0);
            b0.u[1] = *(const U4*)(pb0 + 16);
            const unsigned short* pb1 = &Bsl[(16 + l16) * KP + k0 + half * 8];
            b1.u[0] = *(const U4*)(pb1);
            b1.u[1] = *(const U4*)(pb1 + 16);
            acc0 = __builtin_amdgcn_wmma_f32_16x16x32_bf16(false, af.v, false, b0.v,
                                                           (short)0, acc0, false, false);
            acc1 = __builtin_amdgcn_wmma_f32_16x16x32_bf16(false, af.v, false, b1.v,
                                                           (short)0, acc1, false, false);
        }

        // ---- partials to LDS: C tile elem (M = r+8*half, N = l16 [+16]) ----
#pragma unroll
        for (int r = 0; r < 8; ++r) {
            part[wave * 512 + (r + 8 * half) * 32 + l16]      = acc0[r];
            part[wave * 512 + (r + 8 * half) * 32 + 16 + l16] = acc1[r];
        }
        __syncthreads();

        float rec = 0.0f;
#pragma unroll
        for (int w = 0; w < 8; ++w) rec += part[w * 512 + ub * 32 + ui];

        // ---- fp32 leaky-tanh update for (ub, col) ----
        const long m   = (long)ub * 256 + t;
        const float pre = fminf(fmaxf(ep[m * 2048 + col] + rec, -10.0f), 10.0f);
        hcur = (1.0f - aa) * hcur + aa * tanhf(pre);
        const unsigned short hv = f2bf(hcur);
        hs16[m * 2048 + col]          = hv;   // readout input
        hn[(long)ub * 2048 + col]     = hv;   // next-step broadcast buffer

        __syncthreads();                      // protect `part` reuse
        grid_barrier(cnt, (unsigned)(SCAN_WGS * (t + 1)));
    }
}

// ---------------------------------------------------------------------------
// Launcher
// inputs: 0 x[int 8*256], 1 tok_emb[50257*512], 2 W_in[512*2048],
//         3 W_rec[2048*2048], 4 a[2048], 5 Wb[512*2048], 6 Wa[50257*512],
//         7 bias[50257], 8 h0[2048]
// output: logits fp32 [8*256*50257]
// ---------------------------------------------------------------------------
extern "C" void kernel_launch(void* const* d_in, const int* in_sizes, int n_in,
                              void* d_out, int out_size, void* d_ws, size_t ws_size,
                              hipStream_t stream)
{
    const int*   x       = (const int*)  d_in[0];
    const float* tok_emb = (const float*)d_in[1];
    const float* W_in    = (const float*)d_in[2];
    const float* W_rec   = (const float*)d_in[3];
    const float* a       = (const float*)d_in[4];
    const float* Wb      = (const float*)d_in[5];
    const float* Wa      = (const float*)d_in[6];
    const float* bias    = (const float*)d_in[7];
    const float* h0      = (const float*)d_in[8];
    float*       logits  = (float*)d_out;

    const int  BT  = 8 * 256;                  // 2048 token rows
    const long NWA = (long)50257 * 512;
    const long NWR = (long)2048 * 2048;

    // --- workspace carve-up (all 16B-aligned sizes) ---
    char* ws = (char*)d_ws;
    float* ep = (float*)ws;                       ws += (long)BT * 2048 * 4;    // 16 MB
    unsigned short* hs16   = (unsigned short*)ws; ws += (long)BT * 2048 * 2;    //  8 MB
    unsigned short* emb16  = (unsigned short*)ws; ws += (long)BT * 512  * 2;    //  2 MB
    unsigned short* WinT16 = (unsigned short*)ws; ws += (long)2048 * 512 * 2;   //  2 MB
    unsigned short* Wb16   = (unsigned short*)ws; ws += (long)512 * 2048 * 2;   //  2 MB
    unsigned short* proj16 = (unsigned short*)ws; ws += (long)BT * 512  * 2;    //  2 MB
    unsigned short* Wrec16 = (unsigned short*)ws; ws += NWR * 2;                //  8 MB
    unsigned short* Wa16   = (unsigned short*)ws; ws += NWA * 2;                // 51.5 MB
    unsigned short* hb0    = (unsigned short*)ws; ws += (long)16 * 2048 * 2;    // 64 KB
    unsigned short* hb1    = (unsigned short*)ws; ws += (long)16 * 2048 * 2;    // 64 KB
    unsigned*       cnt    = (unsigned*)ws;       ws += 256;

    // --- one-time operand prep ---
    gather_convert_emb<<<dim3(BT), dim3(128), 0, stream>>>(x, tok_emb, emb16);
    transpose_convert_512x2048<<<dim3(64, 16), dim3(32, 8), 0, stream>>>(W_in, WinT16);
    convert_f32_bf16<<<dim3((int)(((long)512 * 2048 + 1023) / 1024)), dim3(256), 0, stream>>>(
        Wb, Wb16, (long)512 * 2048);
    convert_f32_bf16<<<dim3((int)((NWR + 1023) / 1024)), dim3(256), 0, stream>>>(
        W_rec, Wrec16, NWR);
    convert_f32_bf16<<<dim3((int)((NWA + 1023) / 1024)), dim3(256), 0, stream>>>(
        Wa, Wa16, NWA);
    scan_init<<<dim3(128), dim3(256), 0, stream>>>(h0, hb0, hb1, cnt);

    // --- pipeline ---
    // ep = emb @ W_inT^T   M=2048 N=2048 K=512  -> fp32 for the scan
    gemm_bf16_nt_async<false, false><<<dim3(2048 / BN, BT / BM), 256, 0, stream>>>(
        emb16, 512, WinT16, 512, nullptr, ep, 2048, BT, 2048, 512);

    // reservoir scan (WMMA + async LDS slab + per-step grid barrier)
    const int scanSmem = 32 * KP * 2 + 8 * 512 * 4;   // 147,968 B
    esn_scan_wmma<<<dim3(SCAN_WGS), dim3(256), scanSmem, stream>>>(
        ep, Wrec16, a, h0, hb0, hb1, hs16, cnt);

    // proj = hs @ Wb^T   M=2048 N=512 K=2048  -> bf16
    gemm_bf16_nt_async<false, true><<<dim3(512 / BN, BT / BM), 256, 0, stream>>>(
        hs16, 2048, Wb16, 2048, nullptr, proj16, 512, BT, 512, 2048);

    // logits = proj @ Wa^T + bias   M=2048 N=50257 K=512  -> fp32 out
    gemm_bf16_nt_async<true, false><<<dim3((50257 + BN - 1) / BN, BT / BM), 256, 0, stream>>>(
        proj16, 512, Wa16, 512, bias, logits, 50257, BT, 50257, 512);
}
